// neuralODE_52742198394888
// MI455X (gfx1250) — compile-verified, matrix operand order checked by reading, not run on hardware
//
#include <hip/hip_runtime.h>
#include <math.h>

// ---------------- problem constants ----------------
#define BATCH   1024
#define NNODES  600
#define NEDGE   599          // N-1
#define NSPR    598          // N-2
#define NDOF    2399         // 3*N + E
#define NODAL   1800         // 3*N
#define XSTR    (2*NDOF)     // 4798, row stride of x and out
#define HID     64
#define ROWS    (BATCH*NSPR) // 612352 spring rows
#define TILES16 (ROWS/16)    // 38272 (exact)
#define EPSR    1e-12f

typedef __attribute__((ext_vector_type(16))) _Float16 v16h;
typedef __attribute__((ext_vector_type(8)))  _Float16 v8h;
typedef __attribute__((ext_vector_type(8)))  float    v8f;
typedef float f32x4u __attribute__((ext_vector_type(4), aligned(4))); // 4B-aligned global vec
typedef __attribute__((ext_vector_type(4))) unsigned int u32x4;
typedef __attribute__((ext_vector_type(4))) int          i32x4;
typedef __attribute__((ext_vector_type(8))) int          i32x8;

#if defined(__has_builtin)
#  if __has_builtin(__builtin_amdgcn_tensor_load_to_lds)
#    define HAVE_TDM 1
#  endif
#endif
#ifndef HAVE_TDM
#  define HAVE_TDM 0
#endif

// CDNA5 wave32 WMMA f16 16x16x32, f32 accumulate.
__device__ inline v8f wmma16(v16h a, v16h b, v8f c) {
  return __builtin_amdgcn_wmma_f32_16x16x32_f16(
      false, a, false, b, (short)0, c, false, false);
}

// A/B f16 element->K map (ISA 7.12.2): K = (e&7) + 16*(e>=8) + 8*(lane>=16)
__device__ inline int kmap(int e, int lane) {
  return (e & 7) + ((e >> 3) << 4) + ((lane >> 4) << 3);
}

// Vectorized A/B operand fetch from an f16 LDS matrix stored row-major with
// `stride` halves per row: the K-map makes elems 0..7 / 8..15 two contiguous
// octets -> two ds_load_b128.
__device__ inline v16h load16(const _Float16* buf, int row, int stride,
                              int kbase, int hi) {
  const _Float16* p = buf + row * stride + kbase + 8 * hi;
  v8h a = *(const v8h*)p;
  v8h b = *(const v8h*)(p + 16);
  return __builtin_shufflevector(a, b, 0,1,2,3,4,5,6,7,8,9,10,11,12,13,14,15);
}

// Single-instruction reciprocal (v_rcp_f32), consistent with __expf/__sincosf.
__device__ inline float frcp(float x) { return __builtin_amdgcn_rcpf(x); }

__device__ inline float tanh_fast(float x) {
  x = fminf(15.0f, fmaxf(-15.0f, x));
  float e = __expf(2.0f * x);
  return (e - 1.0f) * frcp(e + 1.0f);
}
__device__ inline float sigmoid_fast(float z) {
  return frcp(1.0f + __expf(-z));
}

__device__ inline void cross3(float ax,float ay,float az,
                              float bx,float by,float bz,
                              float& cx,float& cy,float& cz) {
  cx = ay*bz - az*by; cy = az*bx - ax*bz; cz = ax*by - ay*bx;
}

// ---------------- kernel 0: zero f_el, copy v into out[:, :NDOF] ----------
__global__ void k_prep(const float* __restrict__ x,
                       float* __restrict__ f_el,
                       float* __restrict__ out) {
  int idx = blockIdx.x * 256 + threadIdx.x;
  if (idx >= BATCH * NDOF) return;
  int b = idx / NDOF, r = idx - b * NDOF;
  f_el[idx] = 0.0f;
  out[(long)b * XSTR + r] = x[(long)b * XSTR + NDOF + r];
}

// ---------------- kernel 1: rod strains (B*S rows x 5) ---------------------
__global__ void k_strains(const float* __restrict__ x,
                          const int*   __restrict__ springs,
                          const float* __restrict__ l0e,
                          const float* __restrict__ d1,
                          const float* __restrict__ d2,
                          float* __restrict__ strains) {
  int idx = blockIdx.x * 256 + threadIdx.x;
  if (idx >= ROWS) return;
  int s = idx % NSPR, b = idx / NSPR;
  const int* sp = springs + s * 5;
  int ni = sp[0], nj = sp[1], nk = sp[2], ep = sp[3], en = sp[4];
  const float* q = x + (long)b * XSTR;

  float ep0=q[3*nj]-q[3*ni], ep1=q[3*nj+1]-q[3*ni+1], ep2=q[3*nj+2]-q[3*ni+2];
  float en0=q[3*nk]-q[3*nj], en1=q[3*nk+1]-q[3*nj+1], en2=q[3*nk+2]-q[3*nj+2];
  float thp=q[NODAL+ep], thn=q[NODAL+en];

  float lp = sqrtf(ep0*ep0+ep1*ep1+ep2*ep2);
  float ln = sqrtf(en0*en0+en1*en1+en2*en2);
  float ilp = frcp(lp+EPSR), iln = frcp(ln+EPSR);
  float tp0=ep0*ilp, tp1=ep1*ilp, tp2=ep2*ilp;
  float tn0=en0*iln, tn1=en1*iln, tn2=en2*iln;
  float ichi = frcp(1.0f + tp0*tn0+tp1*tn1+tp2*tn2 + EPSR);
  float cx,cy,cz; cross3(tp0,tp1,tp2, tn0,tn1,tn2, cx,cy,cz);
  float kb0=2.0f*cx*ichi, kb1=2.0f*cy*ichi, kb2=2.0f*cz*ichi;

  float cp, spp, cn, snn;
  __sincosf(thp, &spp, &cp);
  __sincosf(thn, &snn, &cn);
  float d1p0=d1[3*ep],d1p1=d1[3*ep+1],d1p2=d1[3*ep+2];
  float d2p0=d2[3*ep],d2p1=d2[3*ep+1],d2p2=d2[3*ep+2];
  float d1n0=d1[3*en],d1n1=d1[3*en+1],d1n2=d1[3*en+2];
  float d2n0=d2[3*en],d2n1=d2[3*en+1],d2n2=d2[3*en+2];
  float m1p0=cp*d1p0+spp*d2p0, m1p1=cp*d1p1+spp*d2p1, m1p2=cp*d1p2+spp*d2p2;
  float m2p0=-spp*d1p0+cp*d2p0, m2p1=-spp*d1p1+cp*d2p1, m2p2=-spp*d1p2+cp*d2p2;
  float m1n0=cn*d1n0+snn*d2n0, m1n1=cn*d1n1+snn*d2n1, m1n2=cn*d1n2+snn*d2n2;
  float m2n0=-snn*d1n0+cn*d2n0, m2n1=-snn*d1n1+cn*d2n1, m2n2=-snn*d1n2+cn*d2n2;

  float kap1 = 0.5f*(kb0*(m2p0+m2n0)+kb1*(m2p1+m2n1)+kb2*(m2p2+m2n2));
  float kap2 = -0.5f*(kb0*(m1p0+m1n0)+kb1*(m1p1+m1n1)+kb2*(m1p2+m1n2));

  float* o = strains + (long)idx * 5;
  o[0] = lp * frcp(l0e[ep]) - 1.0f;
  o[1] = ln * frcp(l0e[en]) - 1.0f;
  o[2] = kap1;
  o[3] = kap2;
  o[4] = thn - thp;
}

// ---------------- kernel 2: MLP forward + backward via WMMA ----------------
// 4 waves / block, one 16-row tile per wave.  All weights staged in LDS as
// f16 in operand-ready layouts; W2 (16KB f32) arrives via the Tensor Data
// Mover (tensor_load_to_lds) while the other LDS init loops run.
__global__ void __launch_bounds__(128)
k_mlp(const float* __restrict__ strains,
      const float* __restrict__ W1, const float* __restrict__ B1,
      const float* __restrict__ W2, const float* __restrict__ B2,
      const float* __restrict__ W3, const float* __restrict__ B3,
      float* __restrict__ dstrain) {
  __shared__ float sW2stage[HID*HID];                     // TDM landing (f32)
  __shared__ __align__(16) _Float16 sW1b[HID*32];         // fwd B: [n][k<32 pad]
  __shared__ __align__(16) _Float16 sW1r[16*HID];         // bwd B: W1 rows, 5..15 zero
  __shared__ __align__(16) _Float16 sW2b[HID*HID];        // fwd B: col-major W2
  __shared__ __align__(16) _Float16 sW2r[HID*HID];        // bwd B: row-major W2
  __shared__ __align__(16) _Float16 sW3b[16*HID];         // z3  B: row0=W3, rest 0
  __shared__ float sB1[HID], sB2[HID];
  __shared__ float sB3s;
  __shared__ __align__(16) _Float16 hb1[4][16*HID];       // h1 fwd -> dz1 bwd
  __shared__ __align__(16) _Float16 hb2[4][16*HID];       // h2 (kept for bwd)
  __shared__ float zb[4][16];                             // dz3 per row

  const int tid = threadIdx.x;

#if HAVE_TDM
  if (tid < 32) {                       // wave 0 issues async TDM copy of W2
    unsigned long long ga = (unsigned long long)(const void*)W2;
    unsigned lds = (unsigned)(unsigned long long)(void*)&sW2stage[0];
    u32x4 g0;
    g0[0] = 1u;                                          // count=1, user mode
    g0[1] = lds;                                         // lds_addr
    g0[2] = (unsigned)(ga & 0xffffffffu);                // global_addr[31:0]
    g0[3] = (unsigned)((ga >> 32) & 0x1ffffffu) | (2u << 30); // [56:32] | type=2
    i32x8 g1;
    g1[0] = (2 << 16);                                   // data_size = 4B
    g1[1] = (int)((4096u & 0xffffu) << 16);              // tensor_dim0 = 4096 (lo16)
    g1[2] = (int)((4096u >> 16) | (1u << 16));           // dim0 hi | tensor_dim1 = 1
    g1[3] = (int)(4096u << 16);                          // tile_dim0 = 4096
    g1[4] = 1;                                           // tile_dim1=1, tile_dim2=0
    g1[5] = 4096;                                        // tensor_dim0_stride
    g1[6] = 0; g1[7] = 0;
    i32x4 gz = {0,0,0,0};
#if defined(__clang_major__) && __clang_major__ >= 23
    i32x8 z8 = {0,0,0,0,0,0,0,0};
    __builtin_amdgcn_tensor_load_to_lds(g0, g1, gz, gz, z8, 0);
#else
    __builtin_amdgcn_tensor_load_to_lds(g0, g1, gz, gz, 0);
#endif
  }
#else
  for (int t = tid; t < HID*HID; t += 128) sW2stage[t] = W2[t];
#endif

  // overlap: small-weight staging while TDM flies
  for (int t = tid; t < HID*32; t += 128) {
    int n = t >> 5, k = t & 31;
    sW1b[t] = (k < 5) ? (_Float16)W1[k*HID + n] : (_Float16)0.0f;
  }
  for (int t = tid; t < 16*HID; t += 128) {
    int s = t >> 6, h = t & 63;
    sW1r[t] = (s < 5) ? (_Float16)W1[s*HID + h] : (_Float16)0.0f;
    sW3b[t] = (s == 0) ? (_Float16)W3[h] : (_Float16)0.0f;
  }
  if (tid < HID) { sB1[tid]=B1[tid]; sB2[tid]=B2[tid]; }
  if (tid == 0) sB3s = B3[0];

#if HAVE_TDM
  if (tid < 32) __builtin_amdgcn_s_wait_tensorcnt(0);
#endif
  __syncthreads();

  // f16 operand-ready copies of W2 (fwd col-major, bwd row-major)
  for (int t = tid; t < HID*HID; t += 128) {
    int n = t >> 6, k = t & 63;
    sW2b[t] = (_Float16)sW2stage[k*HID + n];
    sW2r[t] = (_Float16)sW2stage[t];
  }
  __syncthreads();

  const int wave = tid >> 5, lane = tid & 31;
  const int lo = lane & 15, hi = lane >> 4;
  long tile = (long)blockIdx.x * 4 + wave;
  if (tile >= TILES16) tile = TILES16 - 1;
  const long row0 = tile * 16;

  _Float16* h1 = hb1[wave];
  _Float16* h2 = hb2[wave];
  float*   dz3 = zb[wave];

  // ----- layer1 fwd: h1 = tanh(strains(16x5,pad32) @ W1 + b1) -----
  v16h a1 = {};
  if (hi == 0) {
    const float* sr = strains + (row0 + lo) * 5;
    #pragma unroll
    for (int e = 0; e < 5; ++e) a1[e] = (_Float16)sr[e];
  }
  #pragma unroll
  for (int nt = 0; nt < 4; ++nt) {
    v16h bt = load16(sW1b, nt*16 + lo, 32, 0, hi);
    v8f c = {};
    c = wmma16(a1, bt, c);
    float bias = sB1[nt*16 + lo];
    #pragma unroll
    for (int r = 0; r < 8; ++r)
      h1[(r + 8*hi)*HID + nt*16 + lo] = (_Float16)tanh_fast(c[r] + bias);
  }
  __syncthreads();

  // ----- layer2 fwd: h2 = tanh(h1 @ W2 + b2) -----
  #pragma unroll
  for (int nt = 0; nt < 4; ++nt) {
    v8f c = {};
    #pragma unroll
    for (int kk = 0; kk < 2; ++kk) {
      v16h a  = load16(h1,   lo,         HID, kk*32, hi);
      v16h bt = load16(sW2b, nt*16 + lo, HID, kk*32, hi);
      c = wmma16(a, bt, c);
    }
    float bias = sB2[nt*16 + lo];
    #pragma unroll
    for (int r = 0; r < 8; ++r)
      h2[(r + 8*hi)*HID + nt*16 + lo] = (_Float16)tanh_fast(c[r] + bias);
  }
  __syncthreads();

  // ----- z3 = h2 @ W3 + b3 (col 0 only);  dz3 = sigmoid(z3) -----
  {
    v8f c = {};
    #pragma unroll
    for (int kk = 0; kk < 2; ++kk) {
      v16h a  = load16(h2,   lo, HID, kk*32, hi);
      v16h bt = load16(sW3b, lo, HID, kk*32, hi);
      c = wmma16(a, bt, c);
    }
    if (lo == 0) {
      #pragma unroll
      for (int r = 0; r < 8; ++r)
        dz3[r + 8*hi] = sigmoid_fast(c[r] + sB3s);
    }
  }
  __syncthreads();

  // ----- bwd: dz2 = dz3*W3*(1-h2^2);  dh1 = dz2 @ W2^T;  dz1 in-place -----
  #pragma unroll
  for (int nt = 0; nt < 4; ++nt) {
    v8f c = {};
    #pragma unroll
    for (int kk = 0; kk < 2; ++kk) {
      const _Float16* hp = h2 + lo*HID + kk*32 + 8*hi;
      v8h h2a = *(const v8h*)hp;
      v8h h2b = *(const v8h*)(hp + 16);
      const _Float16* wp = sW3b + kk*32 + 8*hi;     // row 0 = W3
      v8h w3a = *(const v8h*)wp;
      v8h w3b = *(const v8h*)(wp + 16);
      float d3 = dz3[lo];
      v16h a;
      #pragma unroll
      for (int e = 0; e < 8; ++e) {
        float hv0 = (float)h2a[e], hv1 = (float)h2b[e];
        a[e]   = (_Float16)(d3 * (float)w3a[e] * (1.0f - hv0*hv0));
        a[e+8] = (_Float16)(d3 * (float)w3b[e] * (1.0f - hv1*hv1));
      }
      v16h bt = load16(sW2r, nt*16 + lo, HID, kk*32, hi);   // W2^T
      c = wmma16(a, bt, c);
    }
    #pragma unroll
    for (int r = 0; r < 8; ++r) {
      int mm = r + 8*hi, n = nt*16 + lo;
      float hv = (float)h1[mm*HID + n];
      h1[mm*HID + n] = (_Float16)(c[r] * (1.0f - hv*hv));   // dz1
    }
  }
  __syncthreads();

  // ----- dE/dstrain = dz1 @ W1^T (first 5 cols) -----
  {
    v8f c = {};
    #pragma unroll
    for (int kk = 0; kk < 2; ++kk) {
      v16h a  = load16(h1,   lo, HID, kk*32, hi);
      v16h bt = load16(sW1r, lo, HID, kk*32, hi);
      c = wmma16(a, bt, c);
    }
    if (lo < 5) {
      #pragma unroll
      for (int r = 0; r < 8; ++r)
        dstrain[(row0 + r + 8*hi) * 5 + lo] = c[r];
    }
  }
}

// ---------------- kernel 3: analytic strain-Jacobian^T, scatter forces -----
__global__ void k_scatter(const float* __restrict__ x,
                          const int*   __restrict__ springs,
                          const float* __restrict__ l0e,
                          const float* __restrict__ d1,
                          const float* __restrict__ d2,
                          const float* __restrict__ gs,
                          float* __restrict__ f_el) {
  int idx = blockIdx.x * 256 + threadIdx.x;
  if (idx >= ROWS) return;
  int s = idx % NSPR, b = idx / NSPR;
  const int* sp = springs + s * 5;
  int ni = sp[0], nj = sp[1], nk = sp[2], ep = sp[3], en = sp[4];
  const float* q = x + (long)b * XSTR;

  float ep0=q[3*nj]-q[3*ni], ep1=q[3*nj+1]-q[3*ni+1], ep2=q[3*nj+2]-q[3*ni+2];
  float en0=q[3*nk]-q[3*nj], en1=q[3*nk+1]-q[3*nj+1], en2=q[3*nk+2]-q[3*nj+2];
  float thp=q[NODAL+ep], thn=q[NODAL+en];
  float lp = sqrtf(ep0*ep0+ep1*ep1+ep2*ep2);
  float ln = sqrtf(en0*en0+en1*en1+en2*en2);
  float ilp = frcp(lp+EPSR), iln = frcp(ln+EPSR);
  float tp0=ep0*ilp, tp1=ep1*ilp, tp2=ep2*ilp;
  float tn0=en0*iln, tn1=en1*iln, tn2=en2*iln;
  float ichi = frcp(1.0f + tp0*tn0+tp1*tn1+tp2*tn2 + EPSR);
  float cx,cy,cz; cross3(tp0,tp1,tp2, tn0,tn1,tn2, cx,cy,cz);
  float kb0=2.0f*cx*ichi, kb1=2.0f*cy*ichi, kb2=2.0f*cz*ichi;

  float cp, spp, cn, snn;
  __sincosf(thp, &spp, &cp);
  __sincosf(thn, &snn, &cn);
  float d1p0=d1[3*ep],d1p1=d1[3*ep+1],d1p2=d1[3*ep+2];
  float d2p0=d2[3*ep],d2p1=d2[3*ep+1],d2p2=d2[3*ep+2];
  float d1n0=d1[3*en],d1n1=d1[3*en+1],d1n2=d1[3*en+2];
  float d2n0=d2[3*en],d2n1=d2[3*en+1],d2n2=d2[3*en+2];
  float m1p0=cp*d1p0+spp*d2p0, m1p1=cp*d1p1+spp*d2p1, m1p2=cp*d1p2+spp*d2p2;
  float m2p0=-spp*d1p0+cp*d2p0, m2p1=-spp*d1p1+cp*d2p1, m2p2=-spp*d1p2+cp*d2p2;
  float m1n0=cn*d1n0+snn*d2n0, m1n1=cn*d1n1+snn*d2n1, m1n2=cn*d1n2+snn*d2n2;
  float m2n0=-snn*d1n0+cn*d2n0, m2n1=-snn*d1n1+cn*d2n1, m2n2=-snn*d1n2+cn*d2n2;

  const float* g = gs + (long)idx * 5;
  float g0=g[0], g1=g[1], g2=g[2], g3=g[3], g4=g[4];

  float u0 = 0.5f*(g2*(m2p0+m2n0) - g3*(m1p0+m1n0));
  float u1 = 0.5f*(g2*(m2p1+m2n1) - g3*(m1p1+m1n1));
  float u2 = 0.5f*(g2*(m2p2+m2n2) - g3*(m1p2+m1n2));
  float kbu = kb0*u0 + kb1*u1 + kb2*u2;

  float a0,a1,a2, b0,b1,b2;
  cross3(tn0,tn1,tn2, u0,u1,u2, a0,a1,a2);
  cross3(u0,u1,u2, tp0,tp1,tp2, b0,b1,b2);
  float vtp0 = 2.0f*ichi*a0 - kbu*ichi*tn0;
  float vtp1 = 2.0f*ichi*a1 - kbu*ichi*tn1;
  float vtp2 = 2.0f*ichi*a2 - kbu*ichi*tn2;
  float vtn0 = 2.0f*ichi*b0 - kbu*ichi*tp0;
  float vtn1 = 2.0f*ichi*b1 - kbu*ichi*tp1;
  float vtn2 = 2.0f*ichi*b2 - kbu*ichi*tp2;

  float dptp = tp0*vtp0 + tp1*vtp1 + tp2*vtp2;
  float dntn = tn0*vtn0 + tn1*vtn1 + tn2*vtn2;
  float il0p = frcp(l0e[ep]), il0n = frcp(l0e[en]);
  float gep0 = g0*tp0*il0p + (vtp0 - dptp*tp0)*ilp;
  float gep1 = g0*tp1*il0p + (vtp1 - dptp*tp1)*ilp;
  float gep2 = g0*tp2*il0p + (vtp2 - dptp*tp2)*ilp;
  float gen0 = g1*tn0*il0n + (vtn0 - dntn*tn0)*iln;
  float gen1 = g1*tn1*il0n + (vtn1 - dntn*tn1)*iln;
  float gen2 = g1*tn2*il0n + (vtn2 - dntn*tn2)*iln;

  float kbm1p = kb0*m1p0+kb1*m1p1+kb2*m1p2;
  float kbm2p = kb0*m2p0+kb1*m2p1+kb2*m2p2;
  float kbm1n = kb0*m1n0+kb1*m1n1+kb2*m1n2;
  float kbm2n = kb0*m2n0+kb1*m2n1+kb2*m2n2;
  float gthp = -0.5f*g2*kbm1p - 0.5f*g3*kbm2p - g4;
  float gthn = -0.5f*g2*kbm1n - 0.5f*g3*kbm2n + g4;

  float* fb = f_el + (long)b * NDOF;
  atomicAdd(&fb[3*ni+0],  gep0);
  atomicAdd(&fb[3*ni+1],  gep1);
  atomicAdd(&fb[3*ni+2],  gep2);
  atomicAdd(&fb[3*nj+0],  gen0 - gep0);
  atomicAdd(&fb[3*nj+1],  gen1 - gep1);
  atomicAdd(&fb[3*nj+2],  gen2 - gep2);
  atomicAdd(&fb[3*nk+0], -gen0);
  atomicAdd(&fb[3*nk+1], -gen1);
  atomicAdd(&fb[3*nk+2], -gen2);
  atomicAdd(&fb[NODAL+ep], -gthp);
  atomicAdd(&fb[NODAL+en], -gthn);
}

// ---------------- kernel 4: WMMA GEMM  acc = v @ C^T ; finalize a ----------
// a = (-v@C.T + f_el + f_ext) / diag(M_ff)
__global__ void __launch_bounds__(128)
k_gemm(const float* __restrict__ x, const float* __restrict__ C,
       const float* __restrict__ Mff, const float* __restrict__ f_el,
       const float* __restrict__ f_ext, float* __restrict__ out) {
  const int NT = (NDOF + 15) / 16;            // 150
  const int KSTEPS = (NDOF + 31) / 32;        // 75
  const int tid = threadIdx.x;
  const int wave = tid >> 5, lane = tid & 31;
  const int lo = lane & 15, hi = lane >> 4;
  long wtile = (long)blockIdx.x * 4 + wave;
  const long total = (long)(BATCH/16) * NT;   // 9600
  if (wtile >= total) wtile = total - 1;
  const int mt = (int)(wtile / NT);
  const int nt = (int)(wtile % NT);
  const int n  = nt*16 + lo;                  // output dof / C row
  const int nc = (n < NDOF) ? n : 0;          // clamped row; col never stored
  const long arow = (long)(mt*16 + lo) * XSTR + NDOF;  // v[b, :]
  const float* brow = C + (long)nc * NDOF;

  v8f acc = {};
  for (int kk = 0; kk < KSTEPS - 1; ++kk) {   // full vectorized K-steps
    int kb = kk*32 + 8*hi;
    __builtin_prefetch((const void*)(brow + kb + 64), 0, 0);
    const float* ap = x + arow + kb;
    const float* bp = brow + kb;
    f32x4u av0 = *(const f32x4u*)(ap);
    f32x4u av1 = *(const f32x4u*)(ap + 4);
    f32x4u av2 = *(const f32x4u*)(ap + 16);
    f32x4u av3 = *(const f32x4u*)(ap + 20);
    f32x4u bv0 = *(const f32x4u*)(bp);
    f32x4u bv1 = *(const f32x4u*)(bp + 4);
    f32x4u bv2 = *(const f32x4u*)(bp + 16);
    f32x4u bv3 = *(const f32x4u*)(bp + 20);
    v16h a, bt;
    #pragma unroll
    for (int e = 0; e < 4; ++e) {
      a[e]    = (_Float16)av0[e];  a[e+4]  = (_Float16)av1[e];
      a[e+8]  = (_Float16)av2[e];  a[e+12] = (_Float16)av3[e];
      bt[e]   = (_Float16)bv0[e];  bt[e+4] = (_Float16)bv1[e];
      bt[e+8] = (_Float16)bv2[e];  bt[e+12]= (_Float16)bv3[e];
    }
    acc = wmma16(a, bt, acc);
  }
  { // tail K-step with bounds guard (NDOF = 2399 not a multiple of 32)
    const int kk = KSTEPS - 1;
    v16h a, bt;
    #pragma unroll
    for (int e = 0; e < 16; ++e) {
      int k = kk*32 + kmap(e, lane);
      float av = 0.0f, bv = 0.0f;
      if (k < NDOF) { av = x[arow + k]; bv = brow[k]; }
      a[e]  = (_Float16)av;
      bt[e] = (_Float16)bv;
    }
    acc = wmma16(a, bt, acc);
  }

  if (n < NDOF) {
    float idiag = frcp(Mff[(long)n*NDOF + n]);
    float fe = f_ext[n];
    #pragma unroll
    for (int r = 0; r < 8; ++r) {
      long b = mt*16 + r + 8*hi;
      float aval = (-acc[r] + f_el[b*NDOF + n] + fe) * idiag;
      out[b*XSTR + NDOF + n] = aval;
    }
  }
}

// ---------------- host-side launcher ---------------------------------------
extern "C" void kernel_launch(void* const* d_in, const int* in_sizes, int n_in,
                              void* d_out, int out_size, void* d_ws, size_t ws_size,
                              hipStream_t stream) {
  const float* x      = (const float*)d_in[1];
  const float* W1     = (const float*)d_in[2];
  const float* B1     = (const float*)d_in[3];
  const float* W2     = (const float*)d_in[4];
  const float* B2     = (const float*)d_in[5];
  const float* W3     = (const float*)d_in[6];
  const float* B3     = (const float*)d_in[7];
  const float* C      = (const float*)d_in[8];
  const float* Mff    = (const float*)d_in[9];
  const float* f_ext  = (const float*)d_in[10];
  const float* l0e    = (const float*)d_in[11];
  const float* d1     = (const float*)d_in[12];
  const float* d2     = (const float*)d_in[13];
  const int*   springs= (const int*)d_in[14];
  float* out = (float*)d_out;

  float* ws      = (float*)d_ws;
  float* strains = ws;                          // ROWS*5
  float* dstrain = ws + (size_t)ROWS * 5;       // ROWS*5
  float* f_el    = ws + (size_t)ROWS * 10;      // BATCH*NDOF

  k_prep   <<<(BATCH*NDOF + 255)/256, 256, 0, stream>>>(x, f_el, out);
  k_strains<<<(ROWS + 255)/256,       256, 0, stream>>>(x, springs, l0e, d1, d2, strains);
  k_mlp    <<<TILES16/4,              128, 0, stream>>>(strains, W1, B1, W2, B2, W3, B3, dstrain);
  k_scatter<<<(ROWS + 255)/256,       256, 0, stream>>>(x, springs, l0e, d1, d2, dstrain, f_el);
  const int gemm_tiles = (BATCH/16) * ((NDOF + 15)/16);  // 9600
  k_gemm   <<<gemm_tiles/4,           128, 0, stream>>>(x, C, Mff, f_el, f_ext, out);
}
